// SA_Block_369367187943
// MI455X (gfx1250) — matrix operand order repaired and checked by reading, not compiled, not run on hardware
//
#include <hip/hip_runtime.h>
#include <hip/hip_bf16.h>
#include <math.h>

// CDNA5 (gfx1250) wave32 WMMA implementation of the SA block.
// Memory-bound: k/v intermediates stored bf16 in HBM and moved HBM->LDS with
// global_load_async_to_lds_b128 (ASYNCcnt, double-buffered); all other global
// traffic is 128-bit coalesced + register-pipelined; fragments are
// ds_load_b128; matrix math is v_wmma_f32_16x16x32_bf16 (f32 accum).

typedef __attribute__((ext_vector_type(16))) __bf16 v16bf;
typedef __attribute__((ext_vector_type(8)))  __bf16 v8bf;
typedef __attribute__((ext_vector_type(4)))  __bf16 v4bf;
typedef __attribute__((ext_vector_type(8)))  float  v8f;

#define BM 64
#define BN 64
#define BK 32
#define NTHREADS 512   // 16 waves -> 4x4 grid of 16x16 WMMA tiles
#define APITCH 40      // bf16 pitch for [*][32] tiles (80B: 16B aligned, conflict-free)
#define WPITCH 72      // bf16 pitch for [*][64] tiles (144B: 16B aligned, conflict-free)
#define CPITCH 68      // f32 pitch for 64x64 staging (row float4 aligned + column-safe)

__device__ __forceinline__ v16bf ld_frag(const __bf16* p0, const __bf16* p1) {
  v8bf lo = *(const v8bf*)p0;
  v8bf hi = *(const v8bf*)p1;
  return __builtin_shufflevector(lo, hi, 0, 1, 2, 3, 4, 5, 6, 7,
                                 8, 9, 10, 11, 12, 13, 14, 15);
}
__device__ __forceinline__ v4bf to4bf(float4 v) {
  v4bf w;
  w[0] = (__bf16)v.x; w[1] = (__bf16)v.y; w[2] = (__bf16)v.z; w[3] = (__bf16)v.w;
  return w;
}
// Async HBM->LDS 16B copy (per lane). vdst = LDS byte offset, 64-bit vaddr.
__device__ __forceinline__ void async_ld_b128(unsigned lds_off,
                                              const void* gptr) {
  asm volatile("global_load_async_to_lds_b128 %0, %1, off"
               :: "v"(lds_off), "v"((unsigned long long)(uintptr_t)gptr)
               : "memory");
}
__device__ __forceinline__ void wait_async_le1() {
  asm volatile("s_wait_asynccnt 0x1" ::: "memory");
}
__device__ __forceinline__ void wait_async_0() {
  asm volatile("s_wait_asynccnt 0x0" ::: "memory");
}

// ---------------------------------------------------------------------------
// Generic strided GEMM (templated paths): C = (A (+A2)) * B (+bias) (+Res)
// ---------------------------------------------------------------------------
struct GemmParams {
  const float* A;   long long a_rs, a_cs, a_bs;   int a_mod;
  const float* A2;  long long a2_rs, a2_cs, a2_bs; int a2_mod;
  const float* Bmat; long long b_rs, b_cs, b_bs;  int b_mod;
  float* C;         long long c_rs, c_cs, c_bs;   int c_mod;
  const float* bias;
  const float* Res; long long r_rs, r_cs, r_bs;   int r_mod;
  int K;
};

template<bool A_KC, bool B_KC, bool HAS_A2, bool HAS_BR, bool C_NC>
__global__ __launch_bounds__(NTHREADS)
void gemm_bf16_wmma(GemmParams p) {
  __shared__ __align__(16) __bf16 As[BM][APITCH];
  __shared__ __align__(16) __bf16 Bt[BN][APITCH];
  __shared__ __align__(16) float  Cs[BM][CPITCH];

  const int tid  = threadIdx.x;
  const int wave = tid >> 5;
  const int lane = tid & 31;
  const int wr   = wave >> 2;
  const int wc   = wave & 3;
  const int lm   = lane & 15;
  const int hi   = lane >> 4;
  const int akoff = hi * 8;
  const int bkoff = hi * 16;

  const int m0 = blockIdx.x * BM;
  const int n0 = blockIdx.y * BN;
  const int z  = blockIdx.z;

  const float* Aptr  = p.A + (size_t)(z % p.a_mod) * (size_t)p.a_bs;
  const float* A2ptr = HAS_A2 ? p.A2 + (size_t)(z % p.a2_mod) * (size_t)p.a2_bs : nullptr;
  const float* Bptr  = p.Bmat + (size_t)(z % p.b_mod) * (size_t)p.b_bs;
  float*       Cptr  = p.C + (size_t)(z % p.c_mod) * (size_t)p.c_bs;
  const float* Rptr  = HAS_BR ? p.Res + (size_t)(z % p.r_mod) * (size_t)p.r_bs : nullptr;

  float4 aReg, bReg, a2Reg;
  auto ldA = [&](int k0) {
    if (A_KC) {
      const int r = tid >> 3, c4 = (tid & 7) * 4;
      aReg = *(const float4*)&Aptr[(size_t)(m0 + r) * (size_t)p.a_rs + (size_t)(k0 + c4)];
      if (HAS_A2)
        a2Reg = *(const float4*)&A2ptr[(size_t)(m0 + r) * (size_t)p.a2_rs + (size_t)(k0 + c4)];
    } else {
      const int c = tid >> 4, r4 = (tid & 15) * 4;
      aReg = *(const float4*)&Aptr[(size_t)(m0 + r4) + (size_t)(k0 + c) * (size_t)p.a_cs];
      if (HAS_A2)
        a2Reg = *(const float4*)&A2ptr[(size_t)(m0 + r4) + (size_t)(k0 + c) * (size_t)p.a2_cs];
    }
  };
  auto stA = [&]() {
    float4 v = aReg;
    if (HAS_A2) { v.x += a2Reg.x; v.y += a2Reg.y; v.z += a2Reg.z; v.w += a2Reg.w; }
    if (A_KC) {
      const int r = tid >> 3, c4 = (tid & 7) * 4;
      *(v4bf*)&As[r][c4] = to4bf(v);
    } else {
      const int c = tid >> 4, r4 = (tid & 15) * 4;
      As[r4 + 0][c] = (__bf16)v.x; As[r4 + 1][c] = (__bf16)v.y;
      As[r4 + 2][c] = (__bf16)v.z; As[r4 + 3][c] = (__bf16)v.w;
    }
  };
  auto ldB = [&](int k0) {
    if (B_KC) {
      const int n = tid >> 3, k4 = (tid & 7) * 4;
      bReg = *(const float4*)&Bptr[(size_t)(n0 + n) * (size_t)p.b_cs + (size_t)(k0 + k4)];
    } else {
      const int k = tid >> 4, n4 = (tid & 15) * 4;
      bReg = *(const float4*)&Bptr[(size_t)(k0 + k) * (size_t)p.b_rs + (size_t)(n0 + n4)];
    }
  };
  auto stB = [&]() {
    if (B_KC) {
      const int n = tid >> 3, k4 = (tid & 7) * 4;
      *(v4bf*)&Bt[n][k4] = to4bf(bReg);
    } else {
      const int k = tid >> 4, n4 = (tid & 15) * 4;
      Bt[n4 + 0][k] = (__bf16)bReg.x; Bt[n4 + 1][k] = (__bf16)bReg.y;
      Bt[n4 + 2][k] = (__bf16)bReg.z; Bt[n4 + 3][k] = (__bf16)bReg.w;
    }
  };

  v8f acc = {};
  ldA(0); ldB(0);
  for (int k0 = 0; k0 < p.K; k0 += BK) {
    stA(); stB();
    __syncthreads();
    if (k0 + BK < p.K) { ldA(k0 + BK); ldB(k0 + BK); }
    const int am = wr * 16 + lm;
    const int bn = wc * 16 + lm;
    const v16bf af = ld_frag(&As[am][bkoff], &As[am][bkoff + 8]);
    const v16bf bf = ld_frag(&Bt[bn][bkoff], &Bt[bn][bkoff + 8]);
    acc = __builtin_amdgcn_wmma_f32_16x16x32_bf16(false, af, false, bf,
                                                  (short)0, acc, false, false);
    __syncthreads();
  }

  {
    const int cnl = wc * 16 + lm;
    const float bia = HAS_BR ? p.bias[n0 + cnl] : 0.f;
#pragma unroll
    for (int r = 0; r < 8; ++r) {
      const int cml = wr * 16 + (hi ? 8 + r : r);
      float v = acc[r] + bia;
      if (HAS_BR)
        v += Rptr[(size_t)(m0 + cml) * (size_t)p.r_rs + (size_t)(n0 + cnl) * (size_t)p.r_cs];
      Cs[cml][cnl] = v;
    }
  }
  __syncthreads();
  if (C_NC) {
#pragma unroll
    for (int it = 0; it < 2; ++it) {
      const int idx = tid + it * NTHREADS;
      const int m = idx >> 4, n4 = (idx & 15) * 4;
      *(float4*)&Cptr[(size_t)(m0 + m) * (size_t)p.c_rs + (size_t)(n0 + n4)] =
          *(const float4*)&Cs[m][n4];
    }
  } else {
#pragma unroll
    for (int it = 0; it < 2; ++it) {
      const int idx = tid + it * NTHREADS;
      const int n = idx >> 4, m4 = (idx & 15) * 4;
      float4 v;
      v.x = Cs[m4 + 0][n]; v.y = Cs[m4 + 1][n];
      v.z = Cs[m4 + 2][n]; v.w = Cs[m4 + 3][n];
      *(float4*)&Cptr[(size_t)(n0 + n) * (size_t)p.c_cs + (size_t)(m0 + m4)] = v;
    }
  }
}

// ---------------------------------------------------------------------------
// Fused k & v: per (b,n) GEMM [d x C] * [C x HW], fmaps read once.
// Outputs bf16: kbuf16 [z][d][HW], vbuf_t [z][s][d] (pre-transposed for
// attention phase 4). Both store paths are packed v8bf.
// ---------------------------------------------------------------------------
__global__ __launch_bounds__(NTHREADS)
void kv_gemm_bf16_wmma(const float* __restrict__ Wk, const float* __restrict__ Wv,
                       const float* __restrict__ fm,
                       __bf16* __restrict__ kbuf16, __bf16* __restrict__ vbuf_t,
                       int N, int d, int Cc, int HW) {
  __shared__ __align__(16) __bf16 Ak[64][APITCH];
  __shared__ __align__(16) __bf16 Av[64][APITCH];
  __shared__ __align__(16) __bf16 Bt[64][APITCH];   // Bt[s_local][c]
  __shared__ __align__(16) __bf16 Csb[64][WPITCH];  // bf16 staging

  const int tid  = threadIdx.x;
  const int wave = tid >> 5;
  const int lane = tid & 31;
  const int wr   = wave >> 2;
  const int wc   = wave & 3;
  const int lm   = lane & 15;
  const int hi   = lane >> 4;
  const int akoff = hi * 8;
  const int bkoff = hi * 16;

  const int z  = blockIdx.z;              // b*N + n
  const int n  = z % N;
  const int s0 = blockIdx.y * 64;
  const size_t wbase = (size_t)n * d * Cc;
  const size_t fbase = (size_t)z * Cc * HW;

  float4 kReg, vReg, fReg;
  auto ldT = [&](int k0) {
    const int r = tid >> 3, c4 = (tid & 7) * 4;
    const size_t off = wbase + (size_t)r * Cc + (k0 + c4);
    kReg = *(const float4*)&Wk[off];
    vReg = *(const float4*)&Wv[off];
    const int k = tid >> 4, n4 = (tid & 15) * 4;
    fReg = *(const float4*)&fm[fbase + (size_t)(k0 + k) * HW + s0 + n4];
  };
  auto stT = [&]() {
    const int r = tid >> 3, c4 = (tid & 7) * 4;
    *(v4bf*)&Ak[r][c4] = to4bf(kReg);
    *(v4bf*)&Av[r][c4] = to4bf(vReg);
    const int k = tid >> 4, n4 = (tid & 15) * 4;
    Bt[n4 + 0][k] = (__bf16)fReg.x; Bt[n4 + 1][k] = (__bf16)fReg.y;
    Bt[n4 + 2][k] = (__bf16)fReg.z; Bt[n4 + 3][k] = (__bf16)fReg.w;
  };

  v8f acck = {};
  v8f accv = {};
  ldT(0);
  for (int k0 = 0; k0 < Cc; k0 += BK) {
    stT();
    __syncthreads();
    if (k0 + BK < Cc) ldT(k0 + BK);
    const int am = wr * 16 + lm;
    const int bn = wc * 16 + lm;
    const v16bf afk = ld_frag(&Ak[am][bkoff], &Ak[am][bkoff + 8]);
    const v16bf afv = ld_frag(&Av[am][bkoff], &Av[am][bkoff + 8]);
    const v16bf bf  = ld_frag(&Bt[bn][bkoff], &Bt[bn][bkoff + 8]);
    acck = __builtin_amdgcn_wmma_f32_16x16x32_bf16(false, afk, false, bf,
                                                   (short)0, acck, false, false);
    accv = __builtin_amdgcn_wmma_f32_16x16x32_bf16(false, afv, false, bf,
                                                   (short)0, accv, false, false);
    __syncthreads();
  }

  const int cnl = wc * 16 + lm;           // s_local
  const int d0  = wr * 16 + hi * 8;       // first d row of this lane's group
  // ---- k: stage [d][s], packed row stores to kbuf16[z][d][HW] ----
#pragma unroll
  for (int r = 0; r < 8; ++r) Csb[d0 + r][cnl] = (__bf16)acck[r];
  __syncthreads();
  {
    const int m = tid >> 3, n8 = (tid & 7) * 8;
    *(v8bf*)&kbuf16[(size_t)z * d * HW + (size_t)m * HW + s0 + n8] =
        *(const v8bf*)&Csb[m][n8];
  }
  __syncthreads();
  // ---- v: stage TRANSPOSED [s][d] (one packed b128 per lane), then rows ----
  {
    v8bf w;
#pragma unroll
    for (int r = 0; r < 8; ++r) w[r] = (__bf16)accv[r];
    *(v8bf*)&Csb[cnl][d0] = w;
  }
  __syncthreads();
  {
    const int s = tid >> 3, e8 = (tid & 7) * 8;
    *(v8bf*)&vbuf_t[((size_t)z * HW + s0 + s) * 64 + e8] = *(const v8bf*)&Csb[s][e8];
  }
}

// ---------------------------------------------------------------------------
// Inverse L2 norms.
// ---------------------------------------------------------------------------
__global__ __launch_bounds__(256)
void qnorm_inv(const float* __restrict__ q, float* __restrict__ invq,
               int HW, int Ct) {
  __shared__ float red[256];
  const int blocks_per_b = Ct / 64;
  const int b  = blockIdx.x / blocks_per_b;
  const int j0 = (blockIdx.x % blocks_per_b) * 64;
  const int j  = j0 + (threadIdx.x & 63);
  const int sr = threadIdx.x >> 6;
  const size_t base = (size_t)b * HW * Ct;
  float s = 0.f;
  for (int s0 = sr; s0 < HW; s0 += 4) {
    const float v = q[base + (size_t)s0 * Ct + j];
    s += v * v;
  }
  red[threadIdx.x] = s;
  __syncthreads();
  if (threadIdx.x < 64) {
    const float t = red[threadIdx.x] + red[threadIdx.x + 64] +
                    red[threadIdx.x + 128] + red[threadIdx.x + 192];
    invq[(size_t)b * Ct + j] = 1.0f / fmaxf(sqrtf(t), 1e-12f);
  }
}

__global__ __launch_bounds__(256)
void knorm_inv(const __bf16* __restrict__ k, float* __restrict__ inv, int HW) {
  __shared__ float red[256];
  const size_t base = (size_t)blockIdx.x * HW;
  float s = 0.f;
  for (int i = threadIdx.x * 8; i < HW; i += 256 * 8) {
    const v8bf v = *(const v8bf*)&k[base + i];
#pragma unroll
    for (int j = 0; j < 8; ++j) { const float f = (float)v[j]; s += f * f; }
  }
  red[threadIdx.x] = s;
  __syncthreads();
  for (int w = 128; w > 0; w >>= 1) {
    if ((int)threadIdx.x < w) red[threadIdx.x] += red[threadIdx.x + w];
    __syncthreads();
  }
  if (threadIdx.x == 0) inv[blockIdx.x] = 1.0f / fmaxf(sqrtf(red[0]), 1e-12f);
}

// ---------------------------------------------------------------------------
// Fused attention per (b,n). k and v tiles arrive via double-buffered
// global_load_async_to_lds_b128 (ASYNCcnt). Output written transposed
// [B,HW,Ct] so the FFN reads it with q_in's strides.
// ---------------------------------------------------------------------------
__global__ __launch_bounds__(NTHREADS)
void attn_fused_wmma(const __bf16* __restrict__ kbuf16,
                     const __bf16* __restrict__ vbuf_t,
                     const float* __restrict__ q_in,
                     const float* __restrict__ invk, const float* __restrict__ invq,
                     const float* __restrict__ rescale,
                     float* __restrict__ obuf_t,
                     int Nh, int d, int HW, int Ct) {
  __shared__ __align__(16) __bf16 Ks2[2][64][APITCH]; // Ks[d_i][s_local]
  __shared__ __align__(16) __bf16 Qt[64][APITCH];     // Qt[e][s_local]
  __shared__ __align__(16) float  fstage[64][64];     // scores / store staging
  __shared__ __align__(16) __bf16 attnb[64][WPITCH];  // softmaxed attn [d][e]
  __shared__ __align__(16) __bf16 Vt2[2][64][WPITCH]; // Vt[s_local][e]

  const int z = blockIdx.x;               // b*Nh + n
  const int b = z / Nh;
  const int n = z % Nh;
  const size_t kbase   = (size_t)z * d * HW;
  const size_t vbase   = (size_t)z * HW * 64;
  const size_t qbase   = (size_t)b * HW * Ct + (size_t)n * d;
  const size_t obase_t = (size_t)b * HW * Ct + (size_t)n * d;

  const int tid  = threadIdx.x;
  const int wave = tid >> 5;
  const int lane = tid & 31;
  const int wr   = wave >> 2;
  const int wc   = wave & 3;
  const int lm   = lane & 15;
  const int hi   = lane >> 4;
  const int akoff = hi * 8;
  const int bkoff = hi * 16;

  // ---- Phase 1: Gram[d_i, e] = sum_s k[d_i,s] * q[e,s] ----
  float4 qReg;
  auto ldQ = [&](int s0) {
    const int k = tid >> 4, n4 = (tid & 15) * 4;
    qReg = *(const float4*)&q_in[qbase + (size_t)(s0 + k) * Ct + n4];
  };
  auto stQ = [&]() {
    const int k = tid >> 4, n4 = (tid & 15) * 4;
    Qt[n4 + 0][k] = (__bf16)qReg.x; Qt[n4 + 1][k] = (__bf16)qReg.y;
    Qt[n4 + 2][k] = (__bf16)qReg.z; Qt[n4 + 3][k] = (__bf16)qReg.w;
  };
  auto issueK = [&](int s0, int bufi) {   // 64x32 bf16 tile: 256 lanes x 16B
    if (tid < 256) {
      const int r = tid >> 2, c8 = (tid & 3) * 8;
      async_ld_b128((unsigned)(uintptr_t)&Ks2[bufi][r][c8],
                    &kbuf16[kbase + (size_t)r * HW + s0 + c8]);
    }
  };

  v8f acc = {};
  issueK(0, 0);
  ldQ(0);
  int buf = 0;
  for (int s0 = 0; s0 < HW; s0 += BK) {
    stQ();
    const bool more = (s0 + BK < HW);
    if (more) { ldQ(s0 + BK); issueK(s0 + BK, buf ^ 1); }
    if (more) wait_async_le1(); else wait_async_0();
    __syncthreads();
    const int am = wr * 16 + lm;
    const int bn = wc * 16 + lm;
    const v16bf af = ld_frag(&Ks2[buf][am][bkoff], &Ks2[buf][am][bkoff + 8]);
    const v16bf bf = ld_frag(&Qt[bn][bkoff], &Qt[bn][bkoff + 8]);
    acc = __builtin_amdgcn_wmma_f32_16x16x32_bf16(false, af, false, bf,
                                                  (short)0, acc, false, false);
    __syncthreads();
    buf ^= 1;
  }

  // ---- Phase 2: scale by inverse norms * rescale ----
  {
    const int cnl = wc * 16 + lm;
    const float iq = invq[(size_t)z * d + cnl] * rescale[n];
#pragma unroll
    for (int r = 0; r < 8; ++r) {
      const int cml = wr * 16 + (hi ? 8 + r : r);
      fstage[cml][cnl] = acc[r] * invk[(size_t)z * d + cml] * iq;
    }
  }
  __syncthreads();

  // ---- Phase 3: row softmax, emit bf16 attn (rows d, cols e) ----
  if (tid < 64) {
    float mx = -3.402823466e+38f;
    for (int e = 0; e < 64; ++e) mx = fmaxf(mx, fstage[tid][e]);
    float sum = 0.f;
    for (int e = 0; e < 64; ++e) {
      const float ev = __expf(fstage[tid][e] - mx);
      fstage[tid][e] = ev;
      sum += ev;
    }
    const float is = 1.0f / sum;
    for (int e = 0; e < 64; ++e) attnb[tid][e] = (__bf16)(fstage[tid][e] * is);
  }
  __syncthreads();

  // ---- Phase 4: out^T[s, d] = sum_e v^T[s,e] * attn[d,e] ----
  const int bn = wc * 16 + lm;
  const v16bf bat0 = ld_frag(&attnb[bn][bkoff],      &attnb[bn][bkoff + 8]);
  const v16bf bat1 = ld_frag(&attnb[bn][32 + bkoff], &attnb[bn][32 + bkoff + 8]);

  auto issueV = [&](int sc, int bufi) {   // 64x64 bf16 tile: 512 lanes x 16B
    const int r = tid >> 3, c8 = (tid & 7) * 8;
    async_ld_b128((unsigned)(uintptr_t)&Vt2[bufi][r][c8],
                  &vbuf_t[vbase + (size_t)(sc + r) * 64 + c8]);
  };

  issueV(0, 0);
  buf = 0;
  for (int sc = 0; sc < HW; sc += 64) {
    const bool more = (sc + 64 < HW);
    if (more) issueV(sc + 64, buf ^ 1);
    if (more) wait_async_le1(); else wait_async_0();
    __syncthreads();

    const int am = wr * 16 + lm;           // s_local row of out^T
    const v16bf a0 = ld_frag(&Vt2[buf][am][bkoff],      &Vt2[buf][am][bkoff + 8]);
    const v16bf a1 = ld_frag(&Vt2[buf][am][32 + bkoff], &Vt2[buf][am][32 + bkoff + 8]);
    v8f oc = {};
    oc = __builtin_amdgcn_wmma_f32_16x16x32_bf16(false, a0, false, bat0,
                                                 (short)0, oc, false, false);
    oc = __builtin_amdgcn_wmma_f32_16x16x32_bf16(false, a1, false, bat1,
                                                 (short)0, oc, false, false);
    const int cnl = wc * 16 + lm;          // d column of out^T
#pragma unroll
    for (int r = 0; r < 8; ++r) fstage[wr * 16 + (hi ? 8 + r : r)][cnl] = oc[r];
    __syncthreads();
#pragma unroll
    for (int it = 0; it < 2; ++it) {
      const int idx = tid + it * NTHREADS;
      const int m = idx >> 4, n4 = (idx & 15) * 4;
      *(float4*)&obuf_t[obase_t + (size_t)(sc + m) * Ct + n4] = *(const float4*)&fstage[m][n4];
    }
    __syncthreads();
    buf ^= 1;
  }
}

// ---------------------------------------------------------------------------
// Launch
// ---------------------------------------------------------------------------
extern "C" void kernel_launch(void* const* d_in, const int* in_sizes, int n_in,
                              void* d_out, int out_size, void* d_ws, size_t ws_size,
                              hipStream_t stream) {
  (void)in_sizes; (void)n_in; (void)out_size; (void)ws_size;
  const int B = 16, Nh = 4, Cc = 256, H = 64, W = 64, d = 64;
  const int HW = H * W;        // 4096
  const int Ct = Nh * d;       // 256

  const float* x       = (const float*)d_in[0];
  const float* fmaps   = (const float*)d_in[1];
  const float* Wq      = (const float*)d_in[2];
  const float* Wk      = (const float*)d_in[3];
  const float* Wv      = (const float*)d_in[4];
  const float* rescale = (const float*)d_in[5];
  const float* Wffn    = (const float*)d_in[6];
  const float* bffn    = (const float*)d_in[7];
  float* out = (float*)d_out;

  const size_t tok = (size_t)B * HW * Ct;       // 16,777,216 elems
  float*  qbuf   = (float*)d_ws;                // [B, HW, Ct] f32
  float*  obuf   = qbuf + tok;                  // [B, HW, Ct] f32 (attn out^T)
  __bf16* kbuf16 = (__bf16*)(obuf + tok);       // [B*Nh, d, HW] bf16
  __bf16* vbuf_t = kbuf16 + tok;                // [B*Nh, HW, d] bf16
  float*  invk   = (float*)(vbuf_t + tok);      // [B*Nh*d]
  float*  invq   = invk + (size_t)B * Nh * d;   // [B*Nh*d]

  // ---- Stage 1: q_in[b,s,o] = sum_c x[b,c,s] * Wq[o,c] ----
  {
    GemmParams p{};
    p.A = x;     p.a_rs = 1;  p.a_cs = HW; p.a_bs = (long long)Cc * HW; p.a_mod = B;
    p.A2 = nullptr; p.a2_rs = 0; p.a2_cs = 0; p.a2_bs = 0; p.a2_mod = 1;
    p.Bmat = Wq; p.b_rs = 1;  p.b_cs = Cc; p.b_bs = 0;                  p.b_mod = 1;
    p.C = qbuf;  p.c_rs = Ct; p.c_cs = 1;  p.c_bs = (long long)HW * Ct; p.c_mod = B;
    p.bias = nullptr;
    p.Res = nullptr; p.r_rs = 0; p.r_cs = 0; p.r_bs = 0; p.r_mod = 1;
    p.K = Cc;
    dim3 grid(HW / BM, Ct / BN, B);
    gemm_bf16_wmma<false, true, false, false, true><<<grid, NTHREADS, 0, stream>>>(p);
  }

  // ---- Stage 2: fused k/v grouped 1x1 convs -> bf16 ----
  {
    dim3 grid(1, HW / 64, B * Nh);
    kv_gemm_bf16_wmma<<<grid, NTHREADS, 0, stream>>>(Wk, Wv, fmaps,
                                                     kbuf16, vbuf_t, Nh, d, Cc, HW);
  }

  // ---- Stage 3: inverse L2 norms ----
  qnorm_inv<<<B * (Ct / 64), 256, 0, stream>>>(qbuf, invq, HW, Ct);
  knorm_inv<<<B * Nh * d, 256, 0, stream>>>(kbuf16, invk, HW);

  // ---- Stage 4: fused Gram -> softmax -> out^T (stored [B,HW,Ct]) ----
  attn_fused_wmma<<<B * Nh, NTHREADS, 0, stream>>>(kbuf16, vbuf_t, qbuf,
                                                   invk, invq, rescale,
                                                   obuf, Nh, d, HW, Ct);

  // ---- Stage 5: FFN: y = (out_t + q_in)*Wffn^T + bffn + q_in -> [B,C,H,W] ----
  {
    GemmParams p{};
    p.A = obuf;  p.a_rs  = Ct; p.a_cs  = 1;  p.a_bs  = (long long)HW * Ct; p.a_mod  = B;
    p.A2 = qbuf; p.a2_rs = Ct; p.a2_cs = 1;  p.a2_bs = (long long)HW * Ct; p.a2_mod = B;
    p.Bmat = Wffn; p.b_rs = 1; p.b_cs = Ct;  p.b_bs = 0;                   p.b_mod  = 1;
    p.C = out;   p.c_rs = 1;   p.c_cs = HW;  p.c_bs = (long long)Cc * HW;  p.c_mod  = B;
    p.bias = bffn;
    p.Res = qbuf; p.r_rs = Ct; p.r_cs = 1;   p.r_bs = (long long)HW * Ct;  p.r_mod  = B;
    p.K = Ct;
    dim3 grid(HW / BM, Cc / BN, B);
    gemm_bf16_wmma<true, true, true, true, false><<<grid, NTHREADS, 0, stream>>>(p);
  }
}
